// ShardAttention_27341761806957
// MI455X (gfx1250) — matrix operand order repaired and checked by reading, not compiled, not run on hardware
//
#include <hip/hip_runtime.h>

// ---------------------------------------------------------------------------
// ShardAttention for MI455X (gfx1250, wave32, WMMA 16x16x32 f16->f32)
//
// Fusion:  out = sum_h softmax(Q_h Q_h^T * 0.25) @ (x @ W_h)
//   k_cvt_x   : x f32 -> f16                       (ws: xh, 8 MB)
//   k_tw      : weight f32 -> wT[h][j][k] f16      (ws: wT, 32 MB)
//   per head h:
//     k_gemm_y: yT[b][j][s] = sum_k wT[h][j][k] * xh[b][s][k]   (ws: yT, 8 MB)
//     k_flash : flash attention rows-blocked, V = y_h, accumulates into d_out
//
// Shapes fixed per reference: b=2, t=2048, c=1024, h=16, ck=64.
// ---------------------------------------------------------------------------

typedef _Float16 half_t;
typedef __attribute__((ext_vector_type(16))) _Float16 v16h;
typedef __attribute__((ext_vector_type(8)))  float    v8f;
typedef __attribute__((ext_vector_type(4)))  _Float16 v4h;

union Frag { v16h h; uint4 u[2]; };

// A fragment (16x32 f16, M x K): lane holds row M=Lm; halves i:
//   i=0..7  -> K = k0 + hi*8 + i          (16 contiguous bytes)
//   i=8..15 -> K = k0 + 16 + hi*8 + (i-8) (16 contiguous bytes, +32 B)
__device__ __forceinline__ v16h frag_a(const half_t* base, int ld, int row0,
                                       int k0, int Lm, int hi) {
  const half_t* p = base + (size_t)(row0 + Lm) * ld + k0 + hi * 8;
  Frag f;
  f.u[0] = *(const uint4*)(p);
  f.u[1] = *(const uint4*)(p + 16);
  return f.h;
}

// B fragment (32x16 f16, K x N): lane holds col N=Lm; halves i same K as frag_a
// (i<8: k0+hi*8+i; i>=8: k0+16+hi*8+(i-8)).  Src row-major: Src[n][k], ld halves.
__device__ __forceinline__ v16h frag_b(const half_t* base, int ld, int n0,
                                       int k0, int Lm, int hi) {
  const half_t* p = base + (size_t)(n0 + Lm) * ld + k0 + hi * 8;
  Frag f;
  f.u[0] = *(const uint4*)(p);
  f.u[1] = *(const uint4*)(p + 16);
  return f.h;
}

__device__ __forceinline__ v8f wmma_f16(v16h a, v16h b, v8f c) {
  return __builtin_amdgcn_wmma_f32_16x16x32_f16(false, a, false, b, (short)0, c,
                                                false, false);
}

// --------------------------- x f32 -> f16 ----------------------------------
__global__ __launch_bounds__(256) void k_cvt_x(const float* __restrict__ x,
                                               half_t* __restrict__ xh) {
  const int i = (blockIdx.x * 256 + threadIdx.x) * 4;
  float4 v = *(const float4*)(x + i);
  v4h o = {(half_t)v.x, (half_t)v.y, (half_t)v.z, (half_t)v.w};
  *(v4h*)(xh + i) = o;
}

// ------------------- weight transpose: wT[h][j][k] f16 ----------------------
// wT[h][j][k] = weight[h*1024 + k][j]
__global__ __launch_bounds__(256) void k_tw(const float* __restrict__ w,
                                            half_t* __restrict__ wT) {
  __shared__ half_t tile[32][33];
  const int h = blockIdx.z, kt = blockIdx.x * 32, jt = blockIdx.y * 32;
  const int tx = threadIdx.x, ty = threadIdx.y;
  for (int yy = ty; yy < 32; yy += 8)
    tile[yy][tx] = (half_t)w[(size_t)(h * 1024 + kt + yy) * 1024 + jt + tx];
  __syncthreads();
  for (int yy = ty; yy < 32; yy += 8)
    wT[((size_t)(h * 1024 + jt + yy)) * 1024 + kt + tx] = tile[tx][yy];
}

// ------------------- per-head GEMM: yT[b][j][s] -----------------------------
// D[j][s] = sum_k wT_h[j][k] * xh[s][k];  wave = 16 j-rows x 128 s-cols.
__global__ __launch_bounds__(256) void k_gemm_y(const half_t* __restrict__ xh,
                                                const half_t* __restrict__ wT,
                                                half_t* __restrict__ yT, int h) {
  const int w = threadIdx.x >> 5;
  const int lane = threadIdx.x & 31, Lm = lane & 15, hi = lane >> 4;
  const int sb = blockIdx.x * 128;            // s block (of 4096 = b*t)
  const int j0 = blockIdx.y * 128 + w * 16;   // j rows for this wave
  const half_t* A = wT + (size_t)h * 1024 * 1024;

  v8f acc[8];
#pragma unroll
  for (int st = 0; st < 8; st++) acc[st] = (v8f){0, 0, 0, 0, 0, 0, 0, 0};

  for (int k0 = 0; k0 < 1024; k0 += 32) {
    v16h a = frag_a(A, 1024, j0, k0, Lm, hi);
#pragma unroll
    for (int st = 0; st < 8; st++) {
      v16h bb = frag_b(xh, 1024, sb + st * 16, k0, Lm, hi);
      acc[st] = wmma_f16(a, bb, acc[st]);
    }
  }
  // C/D layout: element r -> (M = j0 + hi*8 + r, N = sb + st*16 + Lm)
#pragma unroll
  for (int st = 0; st < 8; st++) {
    const int s = sb + st * 16 + Lm;
    const int bidx = s >> 11;
    const int t = s & 2047;
    half_t* dst = yT + ((size_t)bidx * 1024) * 2048 + t;
#pragma unroll
    for (int r = 0; r < 8; r++) {
      const int j = j0 + hi * 8 + r;
      dst[(size_t)j * 2048] = (half_t)acc[st][r];
    }
  }
}

// ------------------------- flash attention ----------------------------------
// grid (t/64, b); 512 threads = 16 waves = 4 row-subtiles (rt) x 4 j-slices (jc).
// Wave: 16 rows x 256 output cols (16 acc tiles, 128 VGPRs).
__global__ __launch_bounds__(512) void k_flash(const half_t* __restrict__ xh,
                                               const half_t* __restrict__ yT,
                                               float* __restrict__ out, int h) {
  extern __shared__ char smem[];
  half_t* LDSV = (half_t*)smem;                   // [1024 j][32 s]  64 KB
  half_t* PB   = (half_t*)(smem + 65536);         // [4 rt][16*32]    4 KB
  float*  RED  = (float*)(smem + 65536 + 4096);   // [4 rt][16][4 jc] 1 KB

  const int tid = threadIdx.x;
  const int w = tid >> 5, rt = w >> 2, jc = w & 3;
  const int lane = tid & 31, Lm = lane & 15, hi = lane >> 4;
  const int b = blockIdx.y;
  const int rb = blockIdx.x * 64 + rt * 16;       // first row of this subtile
  const half_t* xb = xh + (size_t)b * 2048 * 1024;
  const half_t* yb = yT + (size_t)b * 1024 * 2048;
  const int hoff = h * 64;
  const float kscale = 0.25f * 1.4426950408889634f;  // /sqrt(h)=0.25, ln2 base

  // Q fragments (rows rb..rb+15, K = head slice of 64)
  v16h q0 = frag_a(xb, 1024, rb, hoff, Lm, hi);
  v16h q1 = frag_a(xb, 1024, rb, hoff + 32, Lm, hi);

  // ---- pass 1: row max; s-range partitioned across jc waves ----
  float m[8];
#pragma unroll
  for (int r = 0; r < 8; r++) m[r] = -3.0e38f;
  for (int s0 = jc * 512; s0 < (jc + 1) * 512; s0 += 16) {
    v16h b0 = frag_b(xb, 1024, s0, hoff, Lm, hi);
    v16h b1 = frag_b(xb, 1024, s0, hoff + 32, Lm, hi);
    v8f c = (v8f){0, 0, 0, 0, 0, 0, 0, 0};
    c = wmma_f16(q0, b0, c);
    c = wmma_f16(q1, b1, c);
#pragma unroll
    for (int r = 0; r < 8; r++) m[r] = fmaxf(m[r], c[r] * kscale);
  }
#pragma unroll
  for (int off = 1; off < 16; off <<= 1)
#pragma unroll
    for (int r = 0; r < 8; r++) m[r] = fmaxf(m[r], __shfl_xor(m[r], off, 32));
  if (Lm == 0) {
#pragma unroll
    for (int r = 0; r < 8; r++) RED[((rt * 16 + hi * 8 + r) << 2) | jc] = m[r];
  }
  __syncthreads();
#pragma unroll
  for (int r = 0; r < 8; r++) {
    const float* q = RED + ((rt * 16 + hi * 8 + r) << 2);
    m[r] = fmaxf(fmaxf(q[0], q[1]), fmaxf(q[2], q[3]));
  }

  // ---- pass 2: recompute S, P = exp2(S*kscale - m), acc += P @ V ----
  float l[8];
#pragma unroll
  for (int r = 0; r < 8; r++) l[r] = 0.f;
  v8f acc[16];
#pragma unroll
  for (int st = 0; st < 16; st++) acc[st] = (v8f){0, 0, 0, 0, 0, 0, 0, 0};

  for (int p = 0; p < 64; p++) {
    const int s0 = p * 32;
    __syncthreads();  // previous iteration's LDSV/PB readers done
    {                 // cooperative V stage: yT rows j, 32 halves (64 B) each
      const int j = tid * 2;
      const uint4* g0 = (const uint4*)(yb + (size_t)j * 2048 + s0);
      const uint4* g1 = (const uint4*)(yb + (size_t)(j + 1) * 2048 + s0);
      uint4* d0 = (uint4*)(LDSV + j * 32);
      uint4* d1 = (uint4*)(LDSV + (j + 1) * 32);
      d0[0] = g0[0]; d0[1] = g0[1]; d0[2] = g0[2]; d0[3] = g0[3];
      d1[0] = g1[0]; d1[1] = g1[1]; d1[2] = g1[2]; d1[3] = g1[3];
    }
    if (jc == (p & 3)) {  // designated wave builds P tile (16x32) for rt
#pragma unroll
      for (int t2 = 0; t2 < 2; t2++) {
        const int sc = s0 + t2 * 16;
        v16h b0 = frag_b(xb, 1024, sc, hoff, Lm, hi);
        v16h b1 = frag_b(xb, 1024, sc, hoff + 32, Lm, hi);
        v8f c = (v8f){0, 0, 0, 0, 0, 0, 0, 0};
        c = wmma_f16(q0, b0, c);
        c = wmma_f16(q1, b1, c);
#pragma unroll
        for (int r = 0; r < 8; r++) {
          float pv = exp2f(c[r] * kscale - m[r]);
          l[r] += pv;
          PB[rt * 512 + (hi * 8 + r) * 32 + t2 * 16 + Lm] = (half_t)pv;
        }
      }
    }
    __syncthreads();  // P + V visible
    v16h a = frag_a(PB + rt * 512, 32, 0, 0, Lm, hi);
#pragma unroll
    for (int st = 0; st < 16; st++) {
      v16h bv = frag_b(LDSV, 32, jc * 256 + st * 16, 0, Lm, hi);
      acc[st] = wmma_f16(a, bv, acc[st]);
    }
  }

  // ---- combine l across lanes and jc waves, normalize, accumulate out ----
#pragma unroll
  for (int off = 1; off < 16; off <<= 1)
#pragma unroll
    for (int r = 0; r < 8; r++) l[r] += __shfl_xor(l[r], off, 32);
  __syncthreads();
  if (Lm == 0) {
#pragma unroll
    for (int r = 0; r < 8; r++) RED[((rt * 16 + hi * 8 + r) << 2) | jc] = l[r];
  }
  __syncthreads();
  float linv[8];
#pragma unroll
  for (int r = 0; r < 8; r++) {
    const float* q = RED + ((rt * 16 + hi * 8 + r) << 2);
    linv[r] = 1.0f / (q[0] + q[1] + q[2] + q[3]);
  }
#pragma unroll
  for (int st = 0; st < 16; st++) {
    const int j = jc * 256 + st * 16 + Lm;
#pragma unroll
    for (int r = 0; r < 8; r++) {
      const int row = rb + hi * 8 + r;
      const size_t idx = ((size_t)(b * 2048 + row)) * 1024 + j;
      const float v = acc[st][r] * linv[r];
      if (h == 0) out[idx] = v;    // first head overwrites poisoned d_out
      else        out[idx] += v;   // rows owned exclusively by this block
    }
  }
}

// ---------------------------------------------------------------------------
extern "C" void kernel_launch(void* const* d_in, const int* in_sizes, int n_in,
                              void* d_out, int out_size, void* d_ws,
                              size_t ws_size, hipStream_t stream) {
  (void)in_sizes; (void)n_in; (void)out_size; (void)ws_size;
  const float* x = (const float*)d_in[0];       // (2, 2048, 1024) f32
  const float* weight = (const float*)d_in[1];  // (16384, 1024) f32
  float* out = (float*)d_out;                   // (2, 2048, 1024) f32

  half_t* xh = (half_t*)d_ws;                   // 4M halves  (8 MB)
  half_t* wT = xh + (size_t)4 * 1024 * 1024;    // 16M halves (32 MB)
  half_t* yT = wT + (size_t)16 * 1024 * 1024;   // 4M halves  (8 MB, per head)

  k_cvt_x<<<4096, 256, 0, stream>>>(x, xh);
  k_tw<<<dim3(32, 32, 16), dim3(32, 8), 0, stream>>>(weight, wT);

  const size_t shmem = 65536 + 4096 + 1024;     // 70,656 B dynamic LDS
  for (int h = 0; h < 16; h++) {
    k_gemm_y<<<dim3(32, 8), 256, 0, stream>>>(xh, wT, yT, h);
    k_flash<<<dim3(32, 2), 512, shmem, stream>>>(xh, yT, out, h);
  }
}
